// HetInfLinkPred_57586921505235
// MI455X (gfx1250) — compile-verified, matrix operand order checked by reading, not run on hardware
//
#include <hip/hip_runtime.h>
#include <hip/hip_bf16.h>
#include <float.h>

// ---------------------------------------------------------------------------
// Problem constants (match reference)
// ---------------------------------------------------------------------------
#define NNODES   50000
#define NEDGES   800000
#define ETOT     (NEDGES + NNODES)   // self loops appended
#define IN_DIM   256
#define HID_DIM  128
#define OUT_DIM  64
#define NEG_SLOPE 0.2f

typedef __attribute__((ext_vector_type(16))) __bf16 v16bf;
typedef __attribute__((ext_vector_type(8)))  float  v8f;

// ---------------------------------------------------------------------------
// Helpers
// ---------------------------------------------------------------------------
__device__ __forceinline__ unsigned int f2bf(float f) {
    unsigned int u = __float_as_uint(f);
    unsigned int r = u + 0x7FFFu + ((u >> 16) & 1u);   // round-to-nearest-even
    return r >> 16;
}

__device__ __forceinline__ uint2 f2bf_pack4(float4 v) {
    return make_uint2(f2bf(v.x) | (f2bf(v.y) << 16),
                      f2bf(v.z) | (f2bf(v.w) << 16));
}

__device__ __forceinline__ void atomicMaxF(float* p, float v) {
    // Sign-aware integer-punned float max (works with -FLT_MAX init).
    if (v >= 0.0f) atomicMax((int*)p, __float_as_int(v));
    else           atomicMin((unsigned int*)p, __float_as_uint(v));
}

__device__ __forceinline__ float eluf(float x) {
    return x > 0.0f ? x : (expf(x) - 1.0f);
}

// ---------------------------------------------------------------------------
// WMMA GEMM: C[M x NCOL] = act(A[M x K](f32) * W[K x NCOL](f32) + bias)
// bf16 operands, f32 accumulate via v_wmma_f32_16x16x32_bf16.
//
// Block = 256 threads (8 waves). Block tile = 128 rows x 32 cols.
// Wave w owns the 16-row strip [m0+16w, +16) and both 16-col subtiles:
// one shared A fragment, two B fragments, two WMMAs per K step.
//
// LDS tiles are K-contiguous with an 80-byte row stride so each
// half-fragment (8 consecutive K per the ISA 16-bit layout) is one aligned
// ds_load_b128, bank-spread across the 64 LDS banks.
//
// OOB rows are *clamped* (not predicated): loads stay in-bounds, garbage
// only reaches LDS rows whose outputs are guarded at the store -> staging
// is branch-free and fully vectorized (global_load_b128 + ds_store_b128).
// ---------------------------------------------------------------------------
#define APAD 40  // row stride in halfwords (80 B = 20 banks)

__global__ __launch_bounds__(256)
void gemm_bf16_wmma(const float* __restrict__ A, int lda,
                    const float* __restrict__ W, int ldw,
                    const float* __restrict__ bias,
                    float* __restrict__ C, int ldc,
                    int M, int K, int NCOL, int relu)
{
    __shared__ unsigned short As[128][APAD]; // 128 rows x 32 K (bf16)
    __shared__ unsigned short Bs[32][APAD];  // transposed: 32 cols x 32 K

    const int wave = threadIdx.x >> 5;
    const int lane = threadIdx.x & 31;
    const int m0   = blockIdx.x * 128;
    const int n0   = blockIdx.y * 32;

    // A staging: thread -> (row, 16-float K chunk), row clamped in-bounds
    const int sr  = threadIdx.x >> 1;
    const int sc  = (threadIdx.x & 1) * 16;
    const int sgr = m0 + sr;
    const float* arow = A + (size_t)((sgr < M) ? sgr : (M - 1)) * lda + sc;

    // B staging: thread -> (col, 4 consecutive K)
    const int bn = threadIdx.x & 31;
    const int bk = (threadIdx.x >> 5) * 4;
    const float* wcol = W + n0 + bn;

    // fragment addressing per ISA 16-bit layout: lane L -> row/col L%16,
    // K halves at khalf and 16+khalf, khalf = (L/16)*8
    const int mrow  = (wave << 4) + (lane & 15);
    const int nc    = lane & 15;
    const int khalf = (lane >> 4) << 3;

    v8f acc0 = {}, acc1 = {};

    for (int k0 = 0; k0 < K; k0 += 32) {
        // ---- stage A: 4x global_load_b128, pack, 2x ds_store_b128 ----
        {
            const float4* src4 = (const float4*)(arow + k0);
            if (k0 + 32 < K) __builtin_prefetch(src4 + 8, 0, 1);  // speculative
            uint2 p0 = f2bf_pack4(src4[0]);
            uint2 p1 = f2bf_pack4(src4[1]);
            uint2 p2 = f2bf_pack4(src4[2]);
            uint2 p3 = f2bf_pack4(src4[3]);
            *(uint4*)&As[sr][sc]     = make_uint4(p0.x, p0.y, p1.x, p1.y);
            *(uint4*)&As[sr][sc + 8] = make_uint4(p2.x, p2.y, p3.x, p3.y);
        }
        // ---- stage B transposed: 4 coalesced b32 loads, 1 ds_store_b64 ----
        {
            float b0 = wcol[(size_t)(k0 + bk + 0) * ldw];
            float b1 = wcol[(size_t)(k0 + bk + 1) * ldw];
            float b2 = wcol[(size_t)(k0 + bk + 2) * ldw];
            float b3 = wcol[(size_t)(k0 + bk + 3) * ldw];
            *(uint2*)&Bs[bn][bk] = make_uint2(f2bf(b0) | (f2bf(b1) << 16),
                                              f2bf(b2) | (f2bf(b3) << 16));
        }
        __syncthreads();

        // ---- fragments: two aligned b128 LDS loads each ----
        union { v16bf v; uint4 q[2]; } fa, fb0, fb1;
        fa.q[0]  = *(const uint4*)&As[mrow][khalf];
        fa.q[1]  = *(const uint4*)&As[mrow][16 + khalf];
        fb0.q[0] = *(const uint4*)&Bs[nc][khalf];
        fb0.q[1] = *(const uint4*)&Bs[nc][16 + khalf];
        fb1.q[0] = *(const uint4*)&Bs[16 + nc][khalf];
        fb1.q[1] = *(const uint4*)&Bs[16 + nc][16 + khalf];

        acc0 = __builtin_amdgcn_wmma_f32_16x16x32_bf16(
                   false, fa.v, false, fb0.v, (short)0, acc0, false, false);
        acc1 = __builtin_amdgcn_wmma_f32_16x16x32_bf16(
                   false, fa.v, false, fb1.v, (short)0, acc1, false, false);
        __syncthreads();
    }

    // ---- store: C/D layout — VGPR r: M = r (lanes 0-15) / r+8 (lanes 16-31)
    const int rbase = (wave << 4) + ((lane >> 4) << 3);
    const int col0  = n0 + nc;
    const int col1  = n0 + 16 + nc;
    const float bia0 = bias ? bias[col0] : 0.0f;
    const float bia1 = bias ? bias[col1] : 0.0f;
    #pragma unroll
    for (int r = 0; r < 8; ++r) {
        int row = m0 + rbase + r;
        if (row < M) {
            float v0 = acc0[r] + bia0;
            float v1 = acc1[r] + bia1;
            if (relu) { v0 = fmaxf(v0, 0.0f); v1 = fmaxf(v1, 0.0f); }
            C[(size_t)row * ldc + col0] = v0;
            C[(size_t)row * ldc + col1] = v1;
        }
    }
}

// ---------------------------------------------------------------------------
// Attention score reduction: s[n,h] = sum_d xh[n,h*D+d] * a[h*D+d]
// H, D compile-time -> divisions become shifts.
// ---------------------------------------------------------------------------
template<int H, int D>
__global__ void score_kernel(const float* __restrict__ xh,
                             const float* __restrict__ asrc,
                             const float* __restrict__ adst,
                             float* __restrict__ ss, float* __restrict__ sd)
{
    int t = blockIdx.x * blockDim.x + threadIdx.x;
    if (t >= NNODES * H) return;
    int n = t / H, h = t % H;
    const float* row = xh + (size_t)n * H * D + h * D;
    const float* as  = asrc + h * D;
    const float* ad  = adst + h * D;
    float a = 0.0f, b = 0.0f;
    #pragma unroll 8
    for (int d = 0; d < D; ++d) { a += row[d] * as[d]; b += row[d] * ad[d]; }
    ss[t] = a; sd[t] = b;
}

// ---------------------------------------------------------------------------
// Buffer fill
// ---------------------------------------------------------------------------
__global__ void fill_kernel(float* __restrict__ p, float v, size_t n) {
    size_t t = (size_t)blockIdx.x * blockDim.x + threadIdx.x;
    if (t < n) p[t] = v;
}

// ---------------------------------------------------------------------------
// Edge pass 1: segment max of leaky_relu(s_src[src]+s_dst[dst]) per dst
// edge id i >= NEDGES is the self loop i-NEDGES.
// ---------------------------------------------------------------------------
template<int H>
__global__ void edge_max_kernel(const int* __restrict__ ei,
                                const float* __restrict__ ss,
                                const float* __restrict__ sd,
                                float* __restrict__ mm)
{
    int i = blockIdx.x * blockDim.x + threadIdx.x;
    if (i >= ETOT) return;
    int s = (i < NEDGES) ? ei[i]          : (i - NEDGES);
    int d = (i < NEDGES) ? ei[NEDGES + i] : (i - NEDGES);
    #pragma unroll
    for (int h = 0; h < H; ++h) {
        float e = ss[s * H + h] + sd[d * H + h];
        e = (e > 0.0f) ? e : e * NEG_SLOPE;
        atomicMaxF(&mm[d * H + h], e);
    }
}

// ---------------------------------------------------------------------------
// Edge pass 2: ee = exp(e - m[dst]); z[dst] += ee
// ---------------------------------------------------------------------------
template<int H>
__global__ void edge_expsum_kernel(const int* __restrict__ ei,
                                   const float* __restrict__ ss,
                                   const float* __restrict__ sd,
                                   const float* __restrict__ mm,
                                   float* __restrict__ ee,
                                   float* __restrict__ zz)
{
    int i = blockIdx.x * blockDim.x + threadIdx.x;
    if (i >= ETOT) return;
    int s = (i < NEDGES) ? ei[i]          : (i - NEDGES);
    int d = (i < NEDGES) ? ei[NEDGES + i] : (i - NEDGES);
    #pragma unroll
    for (int h = 0; h < H; ++h) {
        float e = ss[s * H + h] + sd[d * H + h];
        e = (e > 0.0f) ? e : e * NEG_SLOPE;
        float x = expf(e - mm[d * H + h]);
        ee[(size_t)i * H + h] = x;
        atomicAdd(&zz[d * H + h], x);
    }
}

// ---------------------------------------------------------------------------
// Edge pass 3: out[dst, f] += xh[src, f] * (ee / z[dst,h]) — the bandwidth
// hog. One thread per (edge, feature); F = H*D is a power of two, so the
// edge/feature split is shift/mask. Feature index contiguous => coalesced.
// ---------------------------------------------------------------------------
template<int H, int D>
__global__ void edge_scatter_kernel(const int* __restrict__ ei,
                                    const float* __restrict__ xh,
                                    const float* __restrict__ ee,
                                    const float* __restrict__ zz,
                                    float* __restrict__ out)
{
    constexpr int F = H * D;
    size_t t = (size_t)blockIdx.x * blockDim.x + threadIdx.x;
    if (t >= (size_t)ETOT * F) return;
    int i = (int)(t / F);          // power-of-two -> shift
    int f = (int)(t % F);          // power-of-two -> mask
    int h = f / D;
    int s = (i < NEDGES) ? ei[i]          : (i - NEDGES);
    int d = (i < NEDGES) ? ei[NEDGES + i] : (i - NEDGES);
    float alpha = ee[(size_t)i * H + h] / zz[d * H + h];
    atomicAdd(&out[(size_t)d * F + f], xh[(size_t)s * F + f] * alpha);
}

// ---------------------------------------------------------------------------
// Finalizers
// ---------------------------------------------------------------------------
template<int F>
__global__ void final_concat_elu(float* __restrict__ acc,
                                 const float* __restrict__ b)
{
    size_t t = (size_t)blockIdx.x * blockDim.x + threadIdx.x;
    if (t >= (size_t)NNODES * F) return;
    int f = (int)(t % F);
    acc[t] = eluf(acc[t] + b[f]);
}

template<int D>
__global__ void final_mean2_elu(const float* __restrict__ acc,
                                const float* __restrict__ b,
                                float* __restrict__ out)
{
    size_t t = (size_t)blockIdx.x * blockDim.x + threadIdx.x;
    if (t >= (size_t)NNODES * D) return;
    int n = (int)(t / D), d = (int)(t % D);
    float v = 0.5f * (acc[(size_t)n * 2 * D + d] + acc[(size_t)n * 2 * D + D + d]);
    out[t] = eluf(v + b[d]);
}

template<int D>
__global__ void final_bias(const float* __restrict__ acc,
                           const float* __restrict__ b,
                           float* __restrict__ out)
{
    size_t t = (size_t)blockIdx.x * blockDim.x + threadIdx.x;
    if (t >= (size_t)NNODES * D) return;
    int d = (int)(t % D);
    out[t] = acc[t] + b[d];
}

// ---------------------------------------------------------------------------
// Host-side orchestration
// ---------------------------------------------------------------------------
static inline unsigned cdiv(size_t a, unsigned b) { return (unsigned)((a + b - 1) / b); }

extern "C" void kernel_launch(void* const* d_in, const int* in_sizes, int n_in,
                              void* d_out, int out_size, void* d_ws, size_t ws_size,
                              hipStream_t stream)
{
    const float* x       = (const float*)d_in[0];
    const int*   ei      = (const int*)  d_in[1];
    const float* w1      = (const float*)d_in[2];
    const float* b1      = (const float*)d_in[3];
    const float* w2      = (const float*)d_in[4];
    const float* b2      = (const float*)d_in[5];
    const float* g1_w    = (const float*)d_in[6];
    const float* g1_asrc = (const float*)d_in[7];
    const float* g1_adst = (const float*)d_in[8];
    const float* g1_b    = (const float*)d_in[9];
    const float* g2_w    = (const float*)d_in[10];
    const float* g2_asrc = (const float*)d_in[11];
    const float* g2_adst = (const float*)d_in[12];
    const float* g2_b    = (const float*)d_in[13];
    const float* g3_w    = (const float*)d_in[14];
    const float* g3_asrc = (const float*)d_in[15];
    const float* g3_adst = (const float*)d_in[16];
    const float* g3_b    = (const float*)d_in[17];
    float* out = (float*)d_out;

    // workspace layout (floats)
    float* ws = (float*)d_ws;
    const size_t N = NNODES;
    float* BUF_A = ws;                        // N x 256
    float* BUF_B = BUF_A + N * 256;           // N x 256
    float* XH    = BUF_B + N * 256;           // N x 256
    float* SS    = XH    + N * 256;           // N x 4
    float* SD    = SS    + N * 4;             // N x 4
    float* MM    = SD    + N * 4;             // N x 4
    float* ZZ    = MM    + N * 4;             // N x 4
    float* EE    = ZZ    + N * 4;             // ETOT x 4

    const dim3 gblk(256);
    #define GEMM(A, lda, W, ldw, bias, C, ldc, K, NC, relu)                          \
        gemm_bf16_wmma<<<dim3(cdiv(N,128), (NC)/32), gblk, 0, stream>>>(             \
            A, lda, W, ldw, bias, C, ldc, (int)N, K, NC, relu)

    // ---------------- semantic MLP ----------------
    GEMM(x,     IN_DIM,  w1, HID_DIM, b1, BUF_A, HID_DIM, IN_DIM,  HID_DIM, 1);
    GEMM(BUF_A, HID_DIM, w2, HID_DIM, b2, BUF_B, HID_DIM, HID_DIM, HID_DIM, 1);

    // GAT1: in BUF_B(N,128) -> XH(N,256), H=4,D=64, accum BUF_A, concat+elu
    // GAT2: in BUF_A(N,256) -> XH(N,128), H=2,D=64, accum BUF_B, mean+elu -> BUF_A(N,64)
    // GAT3: in BUF_A(N,64)  -> XH(N,64),  H=1,D=64, accum BUF_B, bias     -> out

    // ===== GAT1 (H=4, D=64, concat) =====
    {
        constexpr int H = 4, D = 64, F = H * D;
        GEMM(BUF_B, HID_DIM, g1_w, F, (const float*)nullptr, XH, F, HID_DIM, F, 0);
        score_kernel<H,D><<<cdiv(N*H,256),256,0,stream>>>(XH, g1_asrc, g1_adst, SS, SD);
        fill_kernel<<<cdiv(N*H,256),256,0,stream>>>(MM, -FLT_MAX, N*H);
        fill_kernel<<<cdiv(N*H,256),256,0,stream>>>(ZZ, 0.0f, N*H);
        fill_kernel<<<cdiv(N*(size_t)F,256),256,0,stream>>>(BUF_A, 0.0f, N*(size_t)F);
        edge_max_kernel<H><<<cdiv(ETOT,256),256,0,stream>>>(ei, SS, SD, MM);
        edge_expsum_kernel<H><<<cdiv(ETOT,256),256,0,stream>>>(ei, SS, SD, MM, EE, ZZ);
        edge_scatter_kernel<H,D><<<cdiv((size_t)ETOT*F,256),256,0,stream>>>(ei, XH, EE, ZZ, BUF_A);
        final_concat_elu<F><<<cdiv(N*(size_t)F,256),256,0,stream>>>(BUF_A, g1_b);
    }

    // ===== GAT2 (H=2, D=64, mean) =====
    {
        constexpr int H = 2, D = 64, F = H * D;
        GEMM(BUF_A, 2*HID_DIM, g2_w, F, (const float*)nullptr, XH, F, 2*HID_DIM, F, 0);
        score_kernel<H,D><<<cdiv(N*H,256),256,0,stream>>>(XH, g2_asrc, g2_adst, SS, SD);
        fill_kernel<<<cdiv(N*H,256),256,0,stream>>>(MM, -FLT_MAX, N*H);
        fill_kernel<<<cdiv(N*H,256),256,0,stream>>>(ZZ, 0.0f, N*H);
        fill_kernel<<<cdiv(N*(size_t)F,256),256,0,stream>>>(BUF_B, 0.0f, N*(size_t)F);
        edge_max_kernel<H><<<cdiv(ETOT,256),256,0,stream>>>(ei, SS, SD, MM);
        edge_expsum_kernel<H><<<cdiv(ETOT,256),256,0,stream>>>(ei, SS, SD, MM, EE, ZZ);
        edge_scatter_kernel<H,D><<<cdiv((size_t)ETOT*F,256),256,0,stream>>>(ei, XH, EE, ZZ, BUF_B);
        final_mean2_elu<D><<<cdiv(N*(size_t)D,256),256,0,stream>>>(BUF_B, g2_b, BUF_A);
    }

    // ===== GAT3 (H=1, D=64, single head) =====
    {
        constexpr int H = 1, D = 64, F = H * D;
        GEMM(BUF_A, OUT_DIM, g3_w, F, (const float*)nullptr, XH, F, OUT_DIM, F, 0);
        score_kernel<H,D><<<cdiv(N*H,256),256,0,stream>>>(XH, g3_asrc, g3_adst, SS, SD);
        fill_kernel<<<cdiv(N*H,256),256,0,stream>>>(MM, -FLT_MAX, N*H);
        fill_kernel<<<cdiv(N*H,256),256,0,stream>>>(ZZ, 0.0f, N*H);
        fill_kernel<<<cdiv(N*(size_t)F,256),256,0,stream>>>(BUF_B, 0.0f, N*(size_t)F);
        edge_max_kernel<H><<<cdiv(ETOT,256),256,0,stream>>>(ei, SS, SD, MM);
        edge_expsum_kernel<H><<<cdiv(ETOT,256),256,0,stream>>>(ei, SS, SD, MM, EE, ZZ);
        edge_scatter_kernel<H,D><<<cdiv((size_t)ETOT*F,256),256,0,stream>>>(ei, XH, EE, ZZ, BUF_B);
        final_bias<D><<<cdiv(N*(size_t)D,256),256,0,stream>>>(BUF_B, g3_b, out);
    }
    #undef GEMM
}
